// MultiHeadAttention_17806934409882
// MI455X (gfx1250) — matrix operand order repaired and checked by reading, not compiled, not run on hardware
//
#include <hip/hip_runtime.h>
#include <hip/hip_bf16.h>

typedef __bf16 bf16;
typedef __attribute__((ext_vector_type(16))) __bf16 bf16x16;
typedef __attribute__((ext_vector_type(8)))  __bf16 bf16x8;
typedef __attribute__((ext_vector_type(4)))  __bf16 bf16x4;
typedef __attribute__((ext_vector_type(8)))  float   v8f;

#define D_MODEL 1024
#define NUM_HEADS 16
#define DEPTH 64
#define BATCH 4
#define SEQ 1024
#define ROWS (BATCH * SEQ)          // 4096

// ---------------------------------------------------------------------------
// WMMA fragment helpers (layouts per CDNA5 ISA 7.12.2, wave32)
// ---------------------------------------------------------------------------

// A-fragment 16x32 bf16: lane (l<16) row M=l holds K {0..7,16..23};
// lanes 16..31 hold K {8..15,24..31}.
__device__ __forceinline__ bf16x16 load_a_bf16(const bf16* __restrict__ base,
                                               int ld, int row, int kt, int half) {
  const bf16* p = base + (size_t)row * ld + kt + (half ? 8 : 0);
  bf16x8 lo = *(const bf16x8*)(p);
  bf16x8 hi = *(const bf16x8*)(p + 16);
  bf16x16 r;
#pragma unroll
  for (int i = 0; i < 8; ++i) { r[i] = lo[i]; r[i + 8] = hi[i]; }
  return r;
}

// A-fragment from f32 source (attn matrix), converted to bf16 in-register.
__device__ __forceinline__ bf16x16 load_a_f32(const float* __restrict__ base,
                                              int ld, int row, int kt, int half) {
  const float* p = base + (size_t)row * ld + kt + (half ? 8 : 0);
  float4 x0 = *(const float4*)(p);
  float4 x1 = *(const float4*)(p + 4);
  float4 x2 = *(const float4*)(p + 16);
  float4 x3 = *(const float4*)(p + 20);
  bf16x16 r;
  r[0]  = (bf16)x0.x; r[1]  = (bf16)x0.y; r[2]  = (bf16)x0.z; r[3]  = (bf16)x0.w;
  r[4]  = (bf16)x1.x; r[5]  = (bf16)x1.y; r[6]  = (bf16)x1.z; r[7]  = (bf16)x1.w;
  r[8]  = (bf16)x2.x; r[9]  = (bf16)x2.y; r[10] = (bf16)x2.z; r[11] = (bf16)x2.w;
  r[12] = (bf16)x3.x; r[13] = (bf16)x3.y; r[14] = (bf16)x3.z; r[15] = (bf16)x3.w;
  return r;
}

// B-fragment 32x16 bf16 from a K-major (B^T row-major [N x K]) buffer.
__device__ __forceinline__ bf16x16 load_b_bf16(const bf16* __restrict__ baseT,
                                               int ld, int col, int kt, int half) {
  const bf16* p = baseT + (size_t)col * ld + kt + (half ? 8 : 0);
  bf16x8 lo = *(const bf16x8*)(p);
  bf16x8 hi = *(const bf16x8*)(p + 16);
  bf16x16 r;
#pragma unroll
  for (int i = 0; i < 8; ++i) { r[i] = lo[i]; r[i + 8] = hi[i]; }
  return r;
}

__device__ __forceinline__ v8f wmma_bf16(bf16x16 a, bf16x16 b, v8f c) {
  return __builtin_amdgcn_wmma_f32_16x16x32_bf16(false, a, false, b,
                                                 (short)0, c, false, false);
}

#define ACC_ZERO {0.f, 0.f, 0.f, 0.f, 0.f, 0.f, 0.f, 0.f}

// ---------------------------------------------------------------------------
// 1) elementwise f32 -> bf16 cast (4 elements / thread)
// ---------------------------------------------------------------------------
__global__ void mha_cast_bf16(const float* __restrict__ in, bf16* __restrict__ out, int n4) {
  int i = blockIdx.x * blockDim.x + threadIdx.x;
  if (i < n4) {
    float4 v = ((const float4*)in)[i];
    bf16x4 o;
    o[0] = (bf16)v.x; o[1] = (bf16)v.y; o[2] = (bf16)v.z; o[3] = (bf16)v.w;
    ((bf16x4*)out)[i] = o;
  }
}

// 2) transpose + cast: WT[n,k] = W[k,n]   (1024x1024)
__global__ void mha_transpose_bf16(const float* __restrict__ W, bf16* __restrict__ WT) {
  int i = blockIdx.x * blockDim.x + threadIdx.x;   // 0 .. 1M-1
  int n = i >> 10, k = i & 1023;
  WT[(size_t)n * 1024 + k] = (bf16)W[(size_t)k * 1024 + n];
}

// ---------------------------------------------------------------------------
// 3) projection GEMM, register-blocked 32x64 per wave:
//    2 A-frags x 4 B-frags -> 8 WMMAs per 32-wide K step.
//    mode 0: store head-split [b,h,s,d] (scaled); mode 2: store [b,h,d,s].
// ---------------------------------------------------------------------------
__global__ void mha_proj_gemm(const bf16* __restrict__ X, const bf16* __restrict__ WT,
                              bf16* __restrict__ out, int mode, float scale) {
  int lane = threadIdx.x & 31;
  int wave = threadIdx.x >> 5;
  int tile = blockIdx.x * 8 + wave;            // 128 x 16 = 2048 macro-tiles
  int row0 = (tile >> 4) * 32;                 // M blocking: 32
  int col0 = (tile & 15) * 64;                 // N blocking: 64
  int half = lane >> 4, nl = lane & 15;

  v8f acc[2][4] = {{ACC_ZERO, ACC_ZERO, ACC_ZERO, ACC_ZERO},
                   {ACC_ZERO, ACC_ZERO, ACC_ZERO, ACC_ZERO}};
#pragma unroll 2
  for (int kt = 0; kt < D_MODEL; kt += 32) {
    if (kt + 32 < D_MODEL) {
      __builtin_prefetch(X + (size_t)(row0 + nl) * D_MODEL + kt + 32, 0, 3);
      __builtin_prefetch(WT + (size_t)(col0 + nl) * D_MODEL + kt + 32, 0, 3);
    }
    bf16x16 a[2], b[4];
#pragma unroll
    for (int i = 0; i < 2; ++i)
      a[i] = load_a_bf16(X, D_MODEL, row0 + i * 16 + nl, kt, half);
#pragma unroll
    for (int j = 0; j < 4; ++j)
      b[j] = load_b_bf16(WT, D_MODEL, col0 + j * 16 + nl, kt, half);
#pragma unroll
    for (int i = 0; i < 2; ++i)
#pragma unroll
      for (int j = 0; j < 4; ++j)
        acc[i][j] = wmma_bf16(a[i], b[j], acc[i][j]);
  }

  int bidx = row0 >> 10;       // 32-row tile never crosses a batch boundary
  int s0   = row0 & 1023;
#pragma unroll
  for (int i = 0; i < 2; ++i) {
#pragma unroll
    for (int j = 0; j < 4; ++j) {
      int n = col0 + j * 16 + nl;
      int h = n >> 6, d = n & 63;
      if (mode == 2) {
        bf16x8 vv;
#pragma unroll
        for (int r = 0; r < 8; ++r) vv[r] = (bf16)(acc[i][j][r] * scale);
        size_t addr = (((size_t)bidx * NUM_HEADS + h) * DEPTH + d) * SEQ
                      + s0 + i * 16 + half * 8;
        *(bf16x8*)(out + addr) = vv;
      } else {
#pragma unroll
        for (int r = 0; r < 8; ++r) {
          int s = s0 + i * 16 + half * 8 + r;
          out[(((size_t)bidx * NUM_HEADS + h) * SEQ + s) * DEPTH + d] =
              (bf16)(acc[i][j][r] * scale);
        }
      }
    }
  }
}

// ---------------------------------------------------------------------------
// 4) logits = Q @ K^T (+mask), 32x64 per wave, K-dim 64 -> 16 WMMAs/wave
// ---------------------------------------------------------------------------
__global__ void mha_logits(const bf16* __restrict__ Q, const bf16* __restrict__ K,
                           const float* __restrict__ mask, float* __restrict__ attn) {
  int lane = threadIdx.x & 31;
  int wave = threadIdx.x >> 5;
  int tile = blockIdx.x * 8 + wave;            // 64 bh * (32 x 16) = 32768
  int bh = tile >> 9;
  int rem = tile & 511;
  int q0 = (rem >> 4) * 32;
  int k0 = (rem & 15) * 64;
  int half = lane >> 4, nl = lane & 15;

  const bf16* Qh = Q + (size_t)bh * SEQ * DEPTH;
  const bf16* Kh = K + (size_t)bh * SEQ * DEPTH;

  v8f acc[2][4] = {{ACC_ZERO, ACC_ZERO, ACC_ZERO, ACC_ZERO},
                   {ACC_ZERO, ACC_ZERO, ACC_ZERO, ACC_ZERO}};
#pragma unroll
  for (int dt = 0; dt < DEPTH; dt += 32) {
    bf16x16 a[2], b[4];
#pragma unroll
    for (int i = 0; i < 2; ++i)
      a[i] = load_a_bf16(Qh, DEPTH, q0 + i * 16 + nl, dt, half);
#pragma unroll
    for (int j = 0; j < 4; ++j)
      b[j] = load_b_bf16(Kh, DEPTH, k0 + j * 16 + nl, dt, half);
#pragma unroll
    for (int i = 0; i < 2; ++i)
#pragma unroll
      for (int j = 0; j < 4; ++j)
        acc[i][j] = wmma_bf16(a[i], b[j], acc[i][j]);
  }

  int bidx = bh >> 4;
  float* ap = attn + (size_t)bh * SEQ * SEQ;
#pragma unroll
  for (int j = 0; j < 4; ++j) {
    float madd = mask[(size_t)bidx * SEQ + k0 + j * 16 + nl] * -1e9f;
#pragma unroll
    for (int i = 0; i < 2; ++i) {
#pragma unroll
      for (int r = 0; r < 8; ++r) {
        int m = q0 + i * 16 + half * 8 + r;
        ap[(size_t)m * SEQ + k0 + j * 16 + nl] = acc[i][j][r] + madd;
      }
    }
  }
}

// ---------------------------------------------------------------------------
// 5) row softmax: one wave32 per 1024-wide row; row lives in 8 float4 regs
// ---------------------------------------------------------------------------
__global__ void mha_softmax(float* __restrict__ attn) {
  int lane = threadIdx.x & 31;
  int wave = threadIdx.x >> 5;
  size_t row = (size_t)blockIdx.x * 8 + wave;
  float* p = attn + row * SEQ;

  float4 v[8];
  float m = -3.402823466e38f;
#pragma unroll
  for (int j = 0; j < 8; ++j) {
    v[j] = *(const float4*)(p + j * 128 + lane * 4);
    m = fmaxf(m, fmaxf(fmaxf(v[j].x, v[j].y), fmaxf(v[j].z, v[j].w)));
  }
#pragma unroll
  for (int off = 16; off >= 1; off >>= 1) m = fmaxf(m, __shfl_xor(m, off, 32));

  float s = 0.f;
#pragma unroll
  for (int j = 0; j < 8; ++j) {
    v[j].x = __expf(v[j].x - m); v[j].y = __expf(v[j].y - m);
    v[j].z = __expf(v[j].z - m); v[j].w = __expf(v[j].w - m);
    s += (v[j].x + v[j].y) + (v[j].z + v[j].w);
  }
#pragma unroll
  for (int off = 16; off >= 1; off >>= 1) s += __shfl_xor(s, off, 32);

  float inv = 1.f / s;
#pragma unroll
  for (int j = 0; j < 8; ++j) {
    v[j].x *= inv; v[j].y *= inv; v[j].z *= inv; v[j].w *= inv;
    *(float4*)(p + j * 128 + lane * 4) = v[j];
  }
}

// ---------------------------------------------------------------------------
// 6) O = attn @ V: 32(s) x 64(full depth) per wave, heads combined on store
// ---------------------------------------------------------------------------
__global__ void mha_av_gemm(const float* __restrict__ attn, const bf16* __restrict__ VT,
                            bf16* __restrict__ O) {
  int lane = threadIdx.x & 31;
  int wave = threadIdx.x >> 5;
  int tile = blockIdx.x * 8 + wave;            // 64 bh * 32 s-tiles = 2048
  int bh = tile >> 5;
  int s0 = (tile & 31) * 32;
  int half = lane >> 4, nl = lane & 15;

  const float* Ah = attn + (size_t)bh * SEQ * SEQ;
  const bf16*  Vh = VT + (size_t)bh * SEQ * DEPTH;   // [d, s] per head

  v8f acc[2][4] = {{ACC_ZERO, ACC_ZERO, ACC_ZERO, ACC_ZERO},
                   {ACC_ZERO, ACC_ZERO, ACC_ZERO, ACC_ZERO}};
#pragma unroll 2
  for (int kt = 0; kt < SEQ; kt += 32) {
    if (kt + 32 < SEQ)
      __builtin_prefetch(Ah + (size_t)(s0 + nl) * SEQ + kt + 32, 0, 3);
    bf16x16 a[2], b[4];
#pragma unroll
    for (int i = 0; i < 2; ++i)
      a[i] = load_a_f32(Ah, SEQ, s0 + i * 16 + nl, kt, half);
#pragma unroll
    for (int j = 0; j < 4; ++j)
      b[j] = load_b_bf16(Vh, SEQ, j * 16 + nl, kt, half);
#pragma unroll
    for (int i = 0; i < 2; ++i)
#pragma unroll
      for (int j = 0; j < 4; ++j)
        acc[i][j] = wmma_bf16(a[i], b[j], acc[i][j]);
  }

  int bidx = bh >> 4, h = bh & 15;
#pragma unroll
  for (int i = 0; i < 2; ++i) {
#pragma unroll
    for (int j = 0; j < 4; ++j) {
#pragma unroll
      for (int r = 0; r < 8; ++r) {
        int s = s0 + i * 16 + half * 8 + r;
        O[((size_t)bidx * SEQ + s) * D_MODEL + h * DEPTH + j * 16 + nl] =
            (bf16)acc[i][j][r];
      }
    }
  }
}

// ---------------------------------------------------------------------------
// 7) out = O @ wo + bo  (f32 output), 32x64 per wave
// ---------------------------------------------------------------------------
__global__ void mha_out_gemm(const bf16* __restrict__ O, const bf16* __restrict__ WOT,
                             const float* __restrict__ bo, float* __restrict__ out) {
  int lane = threadIdx.x & 31;
  int wave = threadIdx.x >> 5;
  int tile = blockIdx.x * 8 + wave;            // 128 x 16 = 2048
  int row0 = (tile >> 4) * 32;
  int col0 = (tile & 15) * 64;
  int half = lane >> 4, nl = lane & 15;

  v8f acc[2][4] = {{ACC_ZERO, ACC_ZERO, ACC_ZERO, ACC_ZERO},
                   {ACC_ZERO, ACC_ZERO, ACC_ZERO, ACC_ZERO}};
#pragma unroll 2
  for (int kt = 0; kt < D_MODEL; kt += 32) {
    if (kt + 32 < D_MODEL) {
      __builtin_prefetch(O + (size_t)(row0 + nl) * D_MODEL + kt + 32, 0, 3);
      __builtin_prefetch(WOT + (size_t)(col0 + nl) * D_MODEL + kt + 32, 0, 3);
    }
    bf16x16 a[2], b[4];
#pragma unroll
    for (int i = 0; i < 2; ++i)
      a[i] = load_a_bf16(O, D_MODEL, row0 + i * 16 + nl, kt, half);
#pragma unroll
    for (int j = 0; j < 4; ++j)
      b[j] = load_b_bf16(WOT, D_MODEL, col0 + j * 16 + nl, kt, half);
#pragma unroll
    for (int i = 0; i < 2; ++i)
#pragma unroll
      for (int j = 0; j < 4; ++j)
        acc[i][j] = wmma_bf16(a[i], b[j], acc[i][j]);
  }

#pragma unroll
  for (int j = 0; j < 4; ++j) {
    float bias = bo[col0 + j * 16 + nl];
#pragma unroll
    for (int i = 0; i < 2; ++i) {
#pragma unroll
      for (int r = 0; r < 8; ++r) {
        int g = row0 + i * 16 + half * 8 + r;
        out[(size_t)g * D_MODEL + col0 + j * 16 + nl] = acc[i][j][r] + bias;
      }
    }
  }
}

// ---------------------------------------------------------------------------
// host launcher
// ---------------------------------------------------------------------------
extern "C" void kernel_launch(void* const* d_in, const int* in_sizes, int n_in,
                              void* d_out, int out_size, void* d_ws, size_t ws_size,
                              hipStream_t stream) {
  const float* q    = (const float*)d_in[0];
  const float* k    = (const float*)d_in[1];
  const float* v    = (const float*)d_in[2];
  const float* mask = (const float*)d_in[3];
  const float* wq   = (const float*)d_in[4];
  const float* wk   = (const float*)d_in[5];
  const float* wv   = (const float*)d_in[6];
  const float* wo   = (const float*)d_in[7];
  const float* bo   = (const float*)d_in[8];

  float* out  = (float*)d_out;
  float* attn = out + (size_t)BATCH * SEQ * D_MODEL;   // output tuple: (out, attn)

  // workspace carve-out (needs 64 MB)
  char* ws = (char*)d_ws;
  size_t off = 0;
  auto carve = [&](size_t bytes) -> void* {
    void* p = ws + off;
    off += (bytes + 255) & ~(size_t)255;
    return p;
  };
  const size_t actBytes = (size_t)ROWS * D_MODEL * sizeof(bf16);   // 8 MB
  const size_t wBytes   = (size_t)D_MODEL * D_MODEL * sizeof(bf16);
  bf16* qbf = (bf16*)carve(actBytes);
  bf16* kbf = (bf16*)carve(actBytes);
  bf16* vbf = (bf16*)carve(actBytes);
  bf16* wqT = (bf16*)carve(wBytes);
  bf16* wkT = (bf16*)carve(wBytes);
  bf16* wvT = (bf16*)carve(wBytes);
  bf16* woT = (bf16*)carve(wBytes);
  bf16* Qh  = (bf16*)carve(actBytes);   // [b,h,s,d], scaled by 1/sqrt(DEPTH)
  bf16* Kh  = (bf16*)carve(actBytes);   // [b,h,s,d]
  bf16* VT  = (bf16*)carve(actBytes);   // [b,h,d,s]
  bf16* O   = (bf16*)carve(actBytes);   // [b,s,h*d]
  (void)ws_size; (void)in_sizes; (void)n_in; (void)out_size;

  dim3 blk(256);

  // casts: 4M elems / tensor, 4 per thread
  mha_cast_bf16<<<4096, blk, 0, stream>>>(q, qbf, ROWS * D_MODEL / 4);
  mha_cast_bf16<<<4096, blk, 0, stream>>>(k, kbf, ROWS * D_MODEL / 4);
  mha_cast_bf16<<<4096, blk, 0, stream>>>(v, vbf, ROWS * D_MODEL / 4);
  // weight transposes (1M elems each)
  mha_transpose_bf16<<<4096, blk, 0, stream>>>(wq, wqT);
  mha_transpose_bf16<<<4096, blk, 0, stream>>>(wk, wkT);
  mha_transpose_bf16<<<4096, blk, 0, stream>>>(wv, wvT);
  mha_transpose_bf16<<<4096, blk, 0, stream>>>(wo, woT);

  const float invsqrtd = 0.125f;   // 1/sqrt(64)
  mha_proj_gemm<<<256, blk, 0, stream>>>(qbf, wqT, Qh, 0, invsqrtd);
  mha_proj_gemm<<<256, blk, 0, stream>>>(kbf, wkT, Kh, 0, 1.0f);
  mha_proj_gemm<<<256, blk, 0, stream>>>(vbf, wvT, VT, 2, 1.0f);

  mha_logits<<<4096, blk, 0, stream>>>(Qh, Kh, mask, attn);
  mha_softmax<<<8192, blk, 0, stream>>>(attn);
  mha_av_gemm<<<256, blk, 0, stream>>>(attn, VT, O);
  mha_out_gemm<<<256, blk, 0, stream>>>(O, woT, bo, out);
}
